// FieldSpaceAttentionModule_45294725104084
// MI455X (gfx1250) — matrix operand order repaired and checked by reading, not compiled, not run on hardware
//
#include <hip/hip_runtime.h>
#include <hip/hip_bf16.h>

// ---------------------------------------------------------------------------
// Types for CDNA5 WMMA (wave32, 16x16x32 f16 -> f32)
// ---------------------------------------------------------------------------
typedef __attribute__((ext_vector_type(16))) _Float16 v16h;
typedef __attribute__((ext_vector_type(8)))  _Float16 h8v;
typedef __attribute__((ext_vector_type(8)))  float    v8f;

union V16U { v16h v; h8v h[2]; };

__device__ __forceinline__ v8f wmma_f16(const V16U& a, const V16U& b, v8f c) {
    return __builtin_amdgcn_wmma_f32_16x16x32_f16(
        /*neg_a=*/false, a.v, /*neg_b=*/false, b.v,
        /*c_mod=*/(short)0, c, /*reuse_a=*/false, /*reuse_b=*/false);
}

#define ZERO8 {0.f,0.f,0.f,0.f,0.f,0.f,0.f,0.f}

static constexpr int N_ROWS = 24576;   // B * N_TOKENS
static constexpr int D_TOK  = 512;

// ---------------------------------------------------------------------------
// Weight prep: f32 [K,N] -> f16 transposed [N,K] (one-time, tiny matrices).
// N is a power of two; pass log2(N).
// ---------------------------------------------------------------------------
__global__ void cvt_transpose_f16(const float* __restrict__ src,
                                  _Float16* __restrict__ dst,
                                  int total, int K, int nbits) {
    int i = blockIdx.x * blockDim.x + threadIdx.x;
    if (i < total) {
        int k = i >> nbits;
        int n = i & ((1 << nbits) - 1);
        dst[(size_t)n * K + k] = (_Float16)src[i];
    }
}

// ---------------------------------------------------------------------------
// LayerNorm over last dim (512), wave per row, f32 in -> f16 out
// ---------------------------------------------------------------------------
__global__ void __launch_bounds__(256)
ln_f32_to_f16(const float* __restrict__ X, const float* __restrict__ g,
              const float* __restrict__ b, _Float16* __restrict__ H, int rows) {
    const int lane = threadIdx.x & 31;
    const int wave = threadIdx.x >> 5;
    const int row  = blockIdx.x * 8 + wave;
    if (row >= rows) return;

    const float* xp = X + (size_t)row * D_TOK;
    float v[16];
    float s = 0.f, ss = 0.f;
#pragma unroll
    for (int i = 0; i < 16; ++i) {
        float x = xp[lane + i * 32];
        v[i] = x; s += x; ss += x * x;
    }
#pragma unroll
    for (int off = 16; off >= 1; off >>= 1) {
        s  += __shfl_xor(s,  off, 32);
        ss += __shfl_xor(ss, off, 32);
    }
    const float mu   = s  * (1.f / 512.f);
    const float var  = ss * (1.f / 512.f) - mu * mu;
    const float rstd = rsqrtf(var + 1e-5f);

    _Float16* hp = H + (size_t)row * D_TOK;
#pragma unroll
    for (int i = 0; i < 16; ++i) {
        int c = lane + i * 32;
        hp[c] = (_Float16)((v[i] - mu) * rstd * g[c] + b[c]);
    }
}

// ---------------------------------------------------------------------------
// WMMA GEMM vs pre-transposed weights:
//   C[M,N](f16,row) = A[M,K](f16,row) @ W (given as Wt[N,K] f16, row-major)
// Block = 256 thr = 8 waves. Tile = 128 rows x 64 cols; each wave 16x64
// (4 accumulators). No LDS, no barriers: both A and B fragments are pairs of
// contiguous 16B global loads (B hits WGP$/L2 - weights are tiny).
// ---------------------------------------------------------------------------
__global__ void __launch_bounds__(256)
gemm_f16_wmma(const _Float16* __restrict__ A, const _Float16* __restrict__ Bt,
              _Float16* __restrict__ C, int K, int lda, int ldc) {
    const int lane = threadIdx.x & 31;
    const int wave = threadIdx.x >> 5;
    const int n0   = blockIdx.x * 64;
    const int m0   = blockIdx.y * 128 + wave * 16;

    // A-fragment (same K order as B): lane<16 -> rows 0..15, koff 0 ; lane>=16 -> koff 16 (contiguous K)
    const int arow  = lane & 15;
    const int akoff = (lane >> 4) * 16;
    // B-fragment (same K order as A): lane&15 = column, halves = contiguous K klo..klo+15
    const int bcol = lane & 15;
    const int bklo = (lane >> 4) * 16;

    v8f acc0 = ZERO8, acc1 = ZERO8, acc2 = ZERO8, acc3 = ZERO8;

    const _Float16* ap  = A  + (size_t)(m0 + arow) * lda + akoff;
    const _Float16* bp0 = Bt + (size_t)(n0 +  0 + bcol) * K + bklo;
    const _Float16* bp1 = Bt + (size_t)(n0 + 16 + bcol) * K + bklo;
    const _Float16* bp2 = Bt + (size_t)(n0 + 32 + bcol) * K + bklo;
    const _Float16* bp3 = Bt + (size_t)(n0 + 48 + bcol) * K + bklo;

    for (int k0 = 0; k0 < K; k0 += 32) {
        V16U a, b0, b1, b2, b3;
        a.h[0]  = *(const h8v*)(ap);
        a.h[1]  = *(const h8v*)(ap + 8);
        b0.h[0] = *(const h8v*)(bp0);
        b0.h[1] = *(const h8v*)(bp0 + 8);
        b1.h[0] = *(const h8v*)(bp1);
        b1.h[1] = *(const h8v*)(bp1 + 8);
        b2.h[0] = *(const h8v*)(bp2);
        b2.h[1] = *(const h8v*)(bp2 + 8);
        b3.h[0] = *(const h8v*)(bp3);
        b3.h[1] = *(const h8v*)(bp3 + 8);

        acc0 = wmma_f16(a, b0, acc0);
        acc1 = wmma_f16(a, b1, acc1);
        acc2 = wmma_f16(a, b2, acc2);
        acc3 = wmma_f16(a, b3, acc3);

        ap += 32; bp0 += 32; bp1 += 32; bp2 += 32; bp3 += 32;
    }

    // C layout: VGPR r -> row r + (lane>>4)*8 ; col = lane&15
    const int crow = (lane >> 4) * 8;
    const int ccol = lane & 15;
    _Float16* cp = C + (size_t)(m0 + crow) * ldc + n0 + ccol;
#pragma unroll
    for (int r = 0; r < 8; ++r) {
        cp[(size_t)r * ldc +  0] = (_Float16)acc0[r];
        cp[(size_t)r * ldc + 16] = (_Float16)acc1[r];
        cp[(size_t)r * ldc + 32] = (_Float16)acc2[r];
        cp[(size_t)r * ldc + 48] = (_Float16)acc3[r];
    }
}

// ---------------------------------------------------------------------------
// Fused attention: per block = one (global seq, head). 512 thr = 16 waves,
// wave owns 16 query rows. QK^T (16 WMMA, K frag direct from global),
// register softmax, P*V (16 WMMA via per-wave LDS re-layout + LDS V^T).
// ---------------------------------------------------------------------------
__global__ void __launch_bounds__(512)
attn_wmma(const _Float16* __restrict__ Q,   // [N_ROWS, 512]
          const _Float16* __restrict__ KV,  // [N_ROWS, 1024] (k | v)
          _Float16* __restrict__ O) {       // [N_ROWS, 512]
    __shared__ _Float16 sVt[32 * 264];      // V^T [j][n], stride 264 (16B aligned)
    __shared__ _Float16 sP[16][16 * 32];    // per-wave P re-layout patch

    const int tid  = threadIdx.x;
    const int lane = tid & 31;
    const int wave = tid >> 5;
    const int head = blockIdx.x;            // 0..15
    const int tok0 = blockIdx.y * 256;      // 96 global sequences
    const int hoff = head * 32;

    // stage V^T: sVt[j][n] = V[n][j]
    for (int e = tid; e < 256 * 32; e += 512) {
        int n = e >> 5, j = e & 31;
        sVt[j * 264 + n] = KV[(size_t)(tok0 + n) * 1024 + 512 + hoff + j];
    }
    __syncthreads();

    const int m0    = wave * 16;
    const int arow  = lane & 15;
    const int akoff = (lane >> 4) * 16;
    const int bcol  = lane & 15;
    const int bklo  = (lane >> 4) * 16;

    // Q fragment (A, 16x32)
    V16U aq;
    {
        const _Float16* qp = Q + (size_t)(tok0 + m0 + arow) * 512 + hoff + akoff;
        aq.h[0] = *(const h8v*)(qp);
        aq.h[1] = *(const h8v*)(qp + 8);
    }

    // scores S[16,256] as 16 WMMA tiles kept in registers
    v8f acc[16];
#pragma unroll
    for (int t = 0; t < 16; ++t) { v8f z = ZERO8; acc[t] = z; }

    const _Float16* kp = KV + (size_t)(tok0 + bcol) * 1024 + hoff + bklo;
#pragma unroll
    for (int t = 0; t < 16; ++t) {
        // B[k,n] = K[n_global, k] -> row-contiguous loads from KV
        V16U bk;
        bk.h[0] = *(const h8v*)(kp);
        bk.h[1] = *(const h8v*)(kp + 8);
        acc[t] = wmma_f16(aq, bk, acc[t]);
        kp += (size_t)16 * 1024;
    }

    // softmax over the 256 columns of each row (scale = 1/sqrt(32))
    const float cexp = 0.17677669529663687f * 1.4426950408889634f; // scale*log2e
    float rmax[8], rinv[8];
#pragma unroll
    for (int r = 0; r < 8; ++r) {
        float m = acc[0][r];
#pragma unroll
        for (int t = 1; t < 16; ++t) m = fmaxf(m, acc[t][r]);
#pragma unroll
        for (int off = 8; off >= 1; off >>= 1) m = fmaxf(m, __shfl_xor(m, off, 32));
        rmax[r] = m;
    }
#pragma unroll
    for (int r = 0; r < 8; ++r) {
        float s = 0.f;
#pragma unroll
        for (int t = 0; t < 16; ++t) {
            float p = exp2f((acc[t][r] - rmax[r]) * cexp);
            acc[t][r] = p; s += p;
        }
#pragma unroll
        for (int off = 8; off >= 1; off >>= 1) s += __shfl_xor(s, off, 32);
        rinv[r] = 1.f / s;
    }

    // O[16,32] = P[16,256] @ V[256,32]
    v8f oacc0 = ZERO8, oacc1 = ZERO8;
    _Float16* pw = &sP[wave][0];
    const int prow = (lane >> 4) * 8;
#pragma unroll
    for (int nc = 0; nc < 8; ++nc) {
        // stage two normalized P tiles (C layout -> LDS row-major 16x32)
#pragma unroll
        for (int half = 0; half < 2; ++half) {
            int t = nc * 2 + half;
#pragma unroll
            for (int r = 0; r < 8; ++r)
                pw[(prow + r) * 32 + half * 16 + (lane & 15)] =
                    (_Float16)(acc[t][r] * rinv[r]);
        }
        V16U ap2, bv0, bv1;
        const _Float16* app = pw + arow * 32 + akoff;
        ap2.h[0] = *(const h8v*)(app);
        ap2.h[1] = *(const h8v*)(app + 8);
        const _Float16* v0 = sVt + (0 + bcol) * 264 + nc * 32 + bklo;
        bv0.h[0] = *(const h8v*)(v0);
        bv0.h[1] = *(const h8v*)(v0 + 8);
        const _Float16* v1 = sVt + (16 + bcol) * 264 + nc * 32 + bklo;
        bv1.h[0] = *(const h8v*)(v1);
        bv1.h[1] = *(const h8v*)(v1 + 8);
        oacc0 = wmma_f16(ap2, bv0, oacc0);
        oacc1 = wmma_f16(ap2, bv1, oacc1);
    }

    // store O
    _Float16* op = O + (size_t)(tok0 + m0 + prow) * 512 + hoff + (lane & 15);
#pragma unroll
    for (int r = 0; r < 8; ++r) {
        op[(size_t)r * 512]      = (_Float16)oacc0[r];
        op[(size_t)r * 512 + 16] = (_Float16)oacc1[r];
    }
}

// ---------------------------------------------------------------------------
// out[i] = base[i] + gamma[i % 512] * (float)delta[i]
// ---------------------------------------------------------------------------
__global__ void axpy_gamma(const float* __restrict__ base,
                           const _Float16* __restrict__ delta,
                           const float* __restrict__ gamma,
                           float* __restrict__ out, int n) {
    int i = blockIdx.x * blockDim.x + threadIdx.x;
    if (i < n) out[i] = base[i] + gamma[i & 511] * (float)delta[i];
}

// ---------------------------------------------------------------------------
// In-place tanh GELU on f16 (matches jax.nn.gelu approximate=True)
// ---------------------------------------------------------------------------
__global__ void gelu_f16(_Float16* __restrict__ h, int n) {
    int i = blockIdx.x * blockDim.x + threadIdx.x;
    if (i < n) {
        float x = (float)h[i];
        float t = tanhf(0.7978845608028654f * (x + 0.044715f * x * x * x));
        h[i] = (_Float16)(0.5f * x * (1.f + t));
    }
}

// ---------------------------------------------------------------------------
// Orchestration
// ---------------------------------------------------------------------------
extern "C" void kernel_launch(void* const* d_in, const int* in_sizes, int n_in,
                              void* d_out, int out_size, void* d_ws, size_t ws_size,
                              hipStream_t stream) {
    const float* x     = (const float*)d_in[0];
    const float* ln1_g = (const float*)d_in[11];
    const float* ln1_b = (const float*)d_in[12];
    const float* ln2_g = (const float*)d_in[13];
    const float* ln2_b = (const float*)d_in[14];
    const float* g_att = (const float*)d_in[15];
    const float* g_mlp = (const float*)d_in[16];
    float* out = (float*)d_out;

    // ---- workspace carve-out (bytes, 256-aligned) ----
    char* ws = (char*)d_ws;
    size_t off = 0;
    auto carve = [&](size_t bytes) -> char* {
        off = (off + 255) & ~(size_t)255;
        char* p = ws + off; off += bytes; return p;
    };
    _Float16* h16   = (_Float16*)carve((size_t)N_ROWS * 512  * 2); // h / h2
    _Float16* t128  = (_Float16*)carve((size_t)N_ROWS * 128  * 2); // rank temps
    _Float16* q16   = (_Float16*)carve((size_t)N_ROWS * 512  * 2); // q, then out2
    _Float16* kv16  = (_Float16*)carve((size_t)N_ROWS * 1024 * 2); // kv, then hid
    _Float16* o16   = (_Float16*)carve((size_t)N_ROWS * 512  * 2); // attn out, then mlp out
    float*    resid = (float*)   carve((size_t)N_ROWS * 512  * 4);
    _Float16* wbuf  = (_Float16*)carve((size_t)851968 * 2);        // transposed f16 weights

    // weight sub-offsets (halves) - stored transposed [N,K]
    _Float16* wQu  = wbuf + 0;
    _Float16* wQv  = wbuf + 65536;
    _Float16* wKVu = wbuf + 131072;
    _Float16* wKVv = wbuf + 196608;
    _Float16* wOu  = wbuf + 327680;
    _Float16* wOv  = wbuf + 393216;
    _Float16* w1u  = wbuf + 458752;
    _Float16* w1v  = wbuf + 524288;
    _Float16* w2u  = wbuf + 655360;
    _Float16* w2v  = wbuf + 786432;

    // ---- convert+transpose weights to f16 [N,K] ----
    struct { int idx; _Float16* dst; int K; int nbits; } wc[10] = {
        {1, wQu,  512,  7},   // [512,128]  -> [128,512]
        {2, wQv,  128,  9},   // [128,512]  -> [512,128]
        {3, wKVu, 512,  7},
        {4, wKVv, 128, 10},   // [128,1024] -> [1024,128]
        {5, wOu,  512,  7},
        {6, wOv,  128,  9},
        {7, w1u,  512,  7},
        {8, w1v,  128, 10},
        {9, w2u, 1024,  7},   // [1024,128] -> [128,1024]
        {10, w2v, 128,  9},
    };
    for (auto& c : wc) {
        int total = c.K << c.nbits;
        cvt_transpose_f16<<<(total + 255) / 256, 256, 0, stream>>>(
            (const float*)d_in[c.idx], c.dst, total, c.K, c.nbits);
    }

    const int nD  = N_ROWS * 512;
    const int n2D = N_ROWS * 1024;
    dim3 blk(256);
    auto ggrid = [](int N) { return dim3(N / 64, N_ROWS / 128); };

    // ---- attention branch ----
    ln_f32_to_f16<<<N_ROWS / 8, blk, 0, stream>>>(x, ln1_g, ln1_b, h16, N_ROWS);

    gemm_f16_wmma<<<ggrid(128),  blk, 0, stream>>>(h16,  wQu,  t128, 512,  512, 128);
    gemm_f16_wmma<<<ggrid(512),  blk, 0, stream>>>(t128, wQv,  q16,  128,  128, 512);
    gemm_f16_wmma<<<ggrid(128),  blk, 0, stream>>>(h16,  wKVu, t128, 512,  512, 128);
    gemm_f16_wmma<<<ggrid(1024), blk, 0, stream>>>(t128, wKVv, kv16, 128,  128, 1024);

    attn_wmma<<<dim3(16, 96), dim3(512), 0, stream>>>(q16, kv16, o16);

    gemm_f16_wmma<<<ggrid(128), blk, 0, stream>>>(o16,  wOu, t128, 512, 512, 128);
    gemm_f16_wmma<<<ggrid(512), blk, 0, stream>>>(t128, wOv, q16,  128, 128, 512);

    axpy_gamma<<<(nD + 255) / 256, blk, 0, stream>>>(x, q16, g_att, resid, nD);

    // ---- MLP branch ----
    ln_f32_to_f16<<<N_ROWS / 8, blk, 0, stream>>>(resid, ln2_g, ln2_b, h16, N_ROWS);

    gemm_f16_wmma<<<ggrid(128),  blk, 0, stream>>>(h16,  w1u, t128, 512, 512, 128);
    gemm_f16_wmma<<<ggrid(1024), blk, 0, stream>>>(t128, w1v, kv16, 128, 128, 1024);
    gelu_f16<<<(n2D + 255) / 256, blk, 0, stream>>>(kv16, n2D);
    gemm_f16_wmma<<<ggrid(128), blk, 0, stream>>>(kv16, w2u, t128, 1024, 1024, 128);
    gemm_f16_wmma<<<ggrid(512), blk, 0, stream>>>(t128, w2v, o16,  128,  128, 512);

    axpy_gamma<<<(nD + 255) / 256, blk, 0, stream>>>(resid, o16, g_mlp, out, nD);
}
